// GNNPrior_33071248179560
// MI455X (gfx1250) — matrix operand order repaired and checked by reading, not compiled, not run on hardware
//
#include <hip/hip_runtime.h>
#include <hip/hip_bf16.h>

typedef __attribute__((ext_vector_type(16))) _Float16 v16h;
typedef __attribute__((ext_vector_type(8)))  _Float16 v8h;
typedef __attribute__((ext_vector_type(4)))  _Float16 v4h;
typedef __attribute__((ext_vector_type(8)))  float    v8f;

#define BN_EPS 1e-5f

__device__ __forceinline__ float softplus_f(float x) {
    // stable softplus: max(x,0) + log1p(exp(-|x|))
    return fmaxf(x, 0.0f) + log1pf(__expf(-fabsf(x)));
}

__device__ __forceinline__ void atomic_add_f32(float* p, float v) {
    __hip_atomic_fetch_add(p, v, __ATOMIC_RELAXED, __HIP_MEMORY_SCOPE_AGENT);
}

// ---------------- elementwise helpers ----------------

__global__ void __launch_bounds__(256) k_zero(float* __restrict__ p, int n) {
    int i = blockIdx.x * blockDim.x + threadIdx.x;
    for (; i < n; i += gridDim.x * blockDim.x) p[i] = 0.0f;
}

__global__ void __launch_bounds__(256) k_copy4(const float4* __restrict__ s,
                                               float4* __restrict__ d, int n4) {
    int i = blockIdx.x * blockDim.x + threadIdx.x;
    for (; i < n4; i += gridDim.x * blockDim.x) d[i] = s[i];
}

// h[n] = node_emb_w[node_type[n]]   (one block of 64 threads per node, float4)
__global__ void __launch_bounds__(64) k_embed(const int* __restrict__ nt,
                                              const float4* __restrict__ emb,
                                              float4* __restrict__ h) {
    int n = blockIdx.x;
    int t = threadIdx.x;                  // 64 * float4 = 256 floats
    h[(size_t)n * 64 + t] = emb[(size_t)nt[n] * 64 + t];
}

// ---------------- message passing ----------------
// agg[dst] += softplus(h[src] + edge_emb_w[edge_type]);  4 edges per block,
// 64 threads/edge, float4 per thread, native f32 atomics.
__global__ void __launch_bounds__(256) k_gconv(const float* __restrict__ h,
                                               const int* __restrict__ src,
                                               const int* __restrict__ dst,
                                               const int* __restrict__ etype,
                                               const float* __restrict__ eemb,
                                               float* __restrict__ agg, int nE) {
    int slot = threadIdx.x >> 6;
    int l    = threadIdx.x & 63;
    int e = blockIdx.x * 4 + slot;
    if (e >= nE) return;
    int s = src[e], d = dst[e], t = etype[e];
    float4 hv = ((const float4*)(h    + (size_t)s * 256))[l];
    float4 ev = ((const float4*)(eemb + (size_t)t * 256))[l];
    float4 m;
    m.x = softplus_f(hv.x + ev.x);
    m.y = softplus_f(hv.y + ev.y);
    m.z = softplus_f(hv.z + ev.z);
    m.w = softplus_f(hv.w + ev.w);
    float* o = agg + (size_t)d * 256 + l * 4;
    atomic_add_f32(o + 0, m.x);
    atomic_add_f32(o + 1, m.y);
    atomic_add_f32(o + 2, m.z);
    atomic_add_f32(o + 3, m.w);
}

// ---------------- batchnorm (training stats) ----------------
// blockDim.x == ncols; thread = feature, rows strided -> coalesced.
__global__ void k_bn_stats(const float* __restrict__ x, float* __restrict__ sum,
                           float* __restrict__ ssq, int nrows, int ncols) {
    int f = threadIdx.x;
    float s = 0.0f, q = 0.0f;
    for (int r = blockIdx.x; r < nrows; r += gridDim.x) {
        float v = x[(size_t)r * ncols + f];
        s += v; q += v * v;
    }
    atomic_add_f32(&sum[f], s);
    atomic_add_f32(&ssq[f], q);
}

__global__ void k_bn_finalize(const float* __restrict__ sum, const float* __restrict__ ssq,
                              const float* __restrict__ gamma, const float* __restrict__ beta,
                              float* __restrict__ scale, float* __restrict__ shift,
                              int nrows, int ncols) {
    int f = blockIdx.x * blockDim.x + threadIdx.x;
    if (f >= ncols) return;
    float inv = 1.0f / (float)nrows;
    float mu  = sum[f] * inv;
    float var = fmaxf(ssq[f] * inv - mu * mu, 0.0f);
    float sc  = gamma[f] * rsqrtf(var + BN_EPS);
    scale[f] = sc;
    shift[f] = beta[f] - mu * sc;
}

// y = maybe_softplus(x*scale + shift), f32 out
__global__ void __launch_bounds__(256) k_bn_act_f32(const float4* __restrict__ x,
                                                    const float* __restrict__ scale,
                                                    const float* __restrict__ shift,
                                                    float4* __restrict__ y,
                                                    int n4, int ncols, int do_act) {
    int i = blockIdx.x * blockDim.x + threadIdx.x;
    for (; i < n4; i += gridDim.x * blockDim.x) {
        int f = (i * 4) & (ncols - 1);      // ncols is power of two (256/128)
        float4 v = x[i];
        v.x = v.x * scale[f + 0] + shift[f + 0];
        v.y = v.y * scale[f + 1] + shift[f + 1];
        v.z = v.z * scale[f + 2] + shift[f + 2];
        v.w = v.w * scale[f + 3] + shift[f + 3];
        if (do_act) {
            v.x = softplus_f(v.x); v.y = softplus_f(v.y);
            v.z = softplus_f(v.z); v.w = softplus_f(v.w);
        }
        y[i] = v;
    }
}

// y = softplus(x*scale + shift) converted to f16 (feeds next WMMA GEMM)
__global__ void __launch_bounds__(256) k_bn_act_f16(const float4* __restrict__ x,
                                                    const float* __restrict__ scale,
                                                    const float* __restrict__ shift,
                                                    v4h* __restrict__ y,
                                                    int n4, int ncols) {
    int i = blockIdx.x * blockDim.x + threadIdx.x;
    for (; i < n4; i += gridDim.x * blockDim.x) {
        int f = (i * 4) & (ncols - 1);
        float4 v = x[i];
        v4h o;
        o[0] = (_Float16)softplus_f(v.x * scale[f + 0] + shift[f + 0]);
        o[1] = (_Float16)softplus_f(v.y * scale[f + 1] + shift[f + 1]);
        o[2] = (_Float16)softplus_f(v.z * scale[f + 2] + shift[f + 2]);
        o[3] = (_Float16)softplus_f(v.w * scale[f + 3] + shift[f + 3]);
        y[i] = o;
    }
}

// ---------------- pooling & feature concat ----------------

__global__ void __launch_bounds__(256) k_pool(const float* __restrict__ h,
                                              const int* __restrict__ batch,
                                              float* __restrict__ hg) {
    int n = blockIdx.x, t = threadIdx.x;
    atomic_add_f32(&hg[(size_t)batch[n] * 256 + t], h[(size_t)n * 256 + t]);
}

// feat16[n] = [ f16(h[n]) | f16(hg[batch[n]]) ]   (row of 512 halfs)
__global__ void __launch_bounds__(256) k_feat(const float* __restrict__ h,
                                              const float* __restrict__ hg,
                                              const int* __restrict__ batch,
                                              _Float16* __restrict__ feat) {
    int n = blockIdx.x, t = threadIdx.x;
    _Float16* row = feat + (size_t)n * 512;
    row[t]       = (_Float16)h[(size_t)n * 256 + t];
    row[256 + t] = (_Float16)hg[(size_t)batch[n] * 256 + t];
}

// Wt[c, k] = f16(W[k, c])   (pre-transpose weights for contiguous B fragments)
__global__ void __launch_bounds__(256) k_wt(const float* __restrict__ W,
                                            _Float16* __restrict__ Wt, int K, int Co) {
    int idx = blockIdx.x * blockDim.x + threadIdx.x;
    if (idx >= K * Co) return;
    int k = idx / Co, c = idx - k * Co;
    Wt[(size_t)c * K + k] = (_Float16)W[idx];
}

// ---------------- WMMA GEMM ----------------
// C[N,Co] = A_f16[N,K] @ Wt_f16[Co,K]^T + bias.
// Block = 8 waves; block tile = 128 rows x 16 cols; B tile staged in LDS.
// splitLat != 0 => final layer: cols [0,LAT) -> d_out, cols [LAT,2LAT) -> d_out+N*LAT.
__global__ void __launch_bounds__(256) k_gemm_wmma(const _Float16* __restrict__ A,
                                                   const _Float16* __restrict__ Wt,
                                                   const float* __restrict__ bias,
                                                   float* __restrict__ C,
                                                   int nrows, int K, int Co, int splitLat) {
    extern __shared__ _Float16 sB[];                 // 16 * K halfs
    const int tid   = threadIdx.x;
    const int tileN = blockIdx.y * 16;

    // cooperative load of the 16-column weight tile (contiguous in Wt)
    const int chunksPerRow = K >> 3;                 // v8h chunks per row
    for (int i = tid; i < 16 * chunksPerRow; i += 256) {
        int j  = i / chunksPerRow;
        int ck = i - j * chunksPerRow;
        *(v8h*)&sB[j * K + ck * 8] = *(const v8h*)&Wt[(size_t)(tileN + j) * K + ck * 8];
    }
    __syncthreads();

    const int wave = tid >> 5;
    const int lane = tid & 31;
    const int m    = lane & 15;
    const int hi   = lane >> 4;
    const int rowBase = blockIdx.x * 128 + wave * 16;

    int arow = rowBase + m;
    if (arow >= nrows) arow = nrows - 1;             // clamp loads; EXEC stays full
    const _Float16* pa = A + (size_t)arow * K + hi * 8;
    const _Float16* pb = sB + m * K + hi * 8;

    v8f acc = {};
    for (int k0 = 0; k0 < K; k0 += 32) {
        __builtin_prefetch(pa + k0 + 128, 0, 1);
        v8h a0 = *(const v8h*)(pa + k0);
        v8h a1 = *(const v8h*)(pa + k0 + 16);
        v8h b0 = *(const v8h*)(pb + k0);
        v8h b1 = *(const v8h*)(pb + k0 + 16);
        v16h a, b;
#pragma unroll
        for (int i = 0; i < 8; ++i) {
            a[i] = a0[i]; a[i + 8] = a1[i];
            b[i] = b0[i]; b[i + 8] = b1[i];
        }
        acc = __builtin_amdgcn_wmma_f32_16x16x32_f16(false, a, false, b,
                                                     (short)0, acc, false, false);
    }

    const int col = tileN + m;
    const float bv = bias[col];
#pragma unroll
    for (int r = 0; r < 8; ++r) {
        int row = rowBase + hi * 8 + r;
        if (row < nrows) {
            float v = acc[r] + bv;
            if (!splitLat) {
                C[(size_t)row * Co + col] = v;
            } else if (col < splitLat) {
                C[(size_t)row * splitLat + col] = v;
            } else {
                C[(size_t)nrows * splitLat + (size_t)row * splitLat + (col - splitLat)] = v;
            }
        }
    }
}

// ---------------- host orchestration ----------------

extern "C" void kernel_launch(void* const* d_in, const int* in_sizes, int n_in,
                              void* d_out, int out_size, void* d_ws, size_t ws_size,
                              hipStream_t stream) {
    const int* node_type  = (const int*)d_in[0];
    const int* edge_type  = (const int*)d_in[1];
    const int* edge_index = (const int*)d_in[2];
    const int* batch      = (const int*)d_in[3];
    const float* node_emb = (const float*)d_in[4];
    const float* edge_emb = (const float*)d_in[5];
    const float* g1 = (const float*)d_in[6],  *b1 = (const float*)d_in[7];
    const float* g2 = (const float*)d_in[8],  *b2 = (const float*)d_in[9];
    const float* g3 = (const float*)d_in[10], *b3 = (const float*)d_in[11];
    const float* go1 = (const float*)d_in[12], *bo1 = (const float*)d_in[13];
    const float* go2 = (const float*)d_in[14], *bo2 = (const float*)d_in[15];
    const float* W1 = (const float*)d_in[16], *B1 = (const float*)d_in[17];
    const float* W2 = (const float*)d_in[18], *B2 = (const float*)d_in[19];
    const float* W3 = (const float*)d_in[20], *B3 = (const float*)d_in[21];

    const int N   = in_sizes[0];
    const int E   = in_sizes[1];
    const int H   = in_sizes[6];          // 256
    const int H2  = H / 2;                // 128
    const int LAT = in_sizes[21] / 2;     // 64
    const int G   = 256;

    const int* src = edge_index;
    const int* dst = edge_index + E;

    // ---- workspace carve ----
    char* p = (char*)d_ws;
    float* bufA = (float*)p;           p += (size_t)N * H * 4;
    float* bufB = (float*)p;           p += (size_t)N * H * 4;
    _Float16* feat16 = (_Float16*)p;   p += (size_t)N * 2 * H * 2;
    _Float16* a1_16  = (_Float16*)p;   p += (size_t)N * H * 2;
    _Float16* a2_16  = (_Float16*)p;   p += (size_t)N * H2 * 2;
    float* hg = (float*)p;             p += (size_t)G * H * 4;
    _Float16* W1t = (_Float16*)p;      p += (size_t)2 * H * H * 2;
    _Float16* W2t = (_Float16*)p;      p += (size_t)H * H2 * 2;
    _Float16* W3t = (_Float16*)p;      p += (size_t)H2 * 2 * LAT * 2;
    float* s_sum   = (float*)p;        p += H * 4;
    float* s_ssq   = (float*)p;        p += H * 4;
    float* s_scale = (float*)p;        p += H * 4;
    float* s_shift = (float*)p;        p += H * 4;

    const int nh4   = (N * H) / 4;
    const int gsELT = (nh4 + 255) / 256;

    // node embedding lookup
    k_embed<<<N, 64, 0, stream>>>(node_type, (const float4*)node_emb, (float4*)bufA);

    // weight convert + transpose (f32 -> f16 [Co,K])
    k_wt<<<(2 * H * H + 255) / 256, 256, 0, stream>>>(W1, W1t, 2 * H, H);
    k_wt<<<(H * H2 + 255) / 256, 256, 0, stream>>>(W2, W2t, H, H2);
    k_wt<<<(H2 * 2 * LAT + 255) / 256, 256, 0, stream>>>(W3, W3t, H2, 2 * LAT);

    // three GNN layers
    const float* gs[3] = {g1, g2, g3};
    const float* bs[3] = {b1, b2, b3};
    for (int layer = 0; layer < 3; ++layer) {
        k_copy4<<<gsELT, 256, 0, stream>>>((const float4*)bufA, (float4*)bufB, nh4);
        k_gconv<<<(E + 3) / 4, 256, 0, stream>>>(bufA, src, dst, edge_type,
                                                 edge_emb, bufB, E);
        k_zero<<<2, 256, 0, stream>>>(s_sum, H);
        k_zero<<<2, 256, 0, stream>>>(s_ssq, H);
        k_bn_stats<<<512, H, 0, stream>>>(bufB, s_sum, s_ssq, N, H);
        k_bn_finalize<<<1, H, 0, stream>>>(s_sum, s_ssq, gs[layer], bs[layer],
                                           s_scale, s_shift, N, H);
        k_bn_act_f32<<<gsELT, 256, 0, stream>>>((const float4*)bufB, s_scale, s_shift,
                                                (float4*)bufA, nh4, H, layer < 2);
    }

    // global pooling + concat features (f16)
    k_zero<<<(G * H + 255) / 256, 256, 0, stream>>>(hg, G * H);
    k_pool<<<N, H, 0, stream>>>(bufA, batch, hg);
    k_feat<<<N, H, 0, stream>>>(bufA, hg, batch, feat16);

    const int gemmGX = (N + 127) / 128;

    // MLP layer 1: [N,512] @ [512,256]
    k_gemm_wmma<<<dim3(gemmGX, H / 16), 256, 16 * (2 * H) * 2, stream>>>(
        feat16, W1t, B1, bufB, N, 2 * H, H, 0);
    k_zero<<<2, 256, 0, stream>>>(s_sum, H);
    k_zero<<<2, 256, 0, stream>>>(s_ssq, H);
    k_bn_stats<<<512, H, 0, stream>>>(bufB, s_sum, s_ssq, N, H);
    k_bn_finalize<<<1, H, 0, stream>>>(s_sum, s_ssq, go1, bo1, s_scale, s_shift, N, H);
    k_bn_act_f16<<<gsELT, 256, 0, stream>>>((const float4*)bufB, s_scale, s_shift,
                                            (v4h*)a1_16, nh4, H);

    // MLP layer 2: [N,256] @ [256,128]
    k_gemm_wmma<<<dim3(gemmGX, H2 / 16), 256, 16 * H * 2, stream>>>(
        a1_16, W2t, B2, bufA, N, H, H2, 0);
    const int nh4b = (N * H2) / 4;
    const int gsELT2 = (nh4b + 255) / 256;
    k_zero<<<1, 256, 0, stream>>>(s_sum, H2);
    k_zero<<<1, 256, 0, stream>>>(s_ssq, H2);
    k_bn_stats<<<512, H2, 0, stream>>>(bufA, s_sum, s_ssq, N, H2);
    k_bn_finalize<<<1, H2, 0, stream>>>(s_sum, s_ssq, go2, bo2, s_scale, s_shift, N, H2);
    k_bn_act_f16<<<gsELT2, 256, 0, stream>>>((const float4*)bufA, s_scale, s_shift,
                                             (v4h*)a2_16, nh4b, H2);

    // MLP layer 3: [N,128] @ [128,128] with split store (mu | logvar)
    k_gemm_wmma<<<dim3(gemmGX, (2 * LAT) / 16), 256, 16 * H2 * 2, stream>>>(
        a2_16, W3t, B3, (float*)d_out, N, H2, 2 * LAT, LAT);
}
